// Cond_Convolutional_5720896438926
// MI455X (gfx1250) — matrix operand order repaired and checked by reading, not compiled, not run on hardware
//
#include <hip/hip_runtime.h>
#include <hip/hip_bf16.h>

// CondConv for MI455X (gfx1250, wave32, WMMA).
// routing -> pack x (NCHW f32 -> NHWC f16) -> pack mixture weights (f16, WMMA
// A-operand lane layout) -> implicit-GEMM conv via v_wmma_f32_16x16x32_f16
// with 32Mx64N register blocking -> BN stats -> BN apply + LeakyReLU.

typedef __attribute__((ext_vector_type(16))) _Float16 v16h;
typedef __attribute__((ext_vector_type(8)))  float    v8f;

#define B_    16
#define CIN   256
#define COUT  512
#define HW    4096   // 64*64
#define NEXP  8
#define KTILES 72    // (9*256) / 32

// ---------------------------------------------------------------- routing ---
__global__ __launch_bounds__(256) void routing_kernel(
    const float* __restrict__ x, const float* __restrict__ fc_w,
    const float* __restrict__ fc_b, float* __restrict__ routing) {
  int b = blockIdx.x;
  int wave = threadIdx.x >> 5, lane = threadIdx.x & 31;
  __shared__ float means[CIN];
  const float* xb = x + (size_t)b * CIN * HW;
  for (int c = wave; c < CIN; c += 8) {
    const float* p = xb + (size_t)c * HW;
    float s = 0.f;
    for (int i = lane; i < HW; i += 32) s += p[i];
    for (int off = 16; off > 0; off >>= 1) s += __shfl_down(s, off, 32);
    if (lane == 0) means[c] = s * (1.0f / (float)HW);
  }
  __syncthreads();
  if (threadIdx.x < NEXP) {
    float a = fc_b[threadIdx.x];
    const float* wr = fc_w + threadIdx.x * CIN;
    for (int c = 0; c < CIN; ++c) a += means[c] * wr[c];
    routing[b * NEXP + threadIdx.x] = 1.0f / (1.0f + __expf(-a));
  }
}

// ------------------------------------------------------- pack x to NHWC f16 --
__global__ __launch_bounds__(256) void pack_x_kernel(
    const float* __restrict__ x, _Float16* __restrict__ xp) {
  __shared__ _Float16 tile[64][258];   // +2 halves pad -> conflict-free
  int b = blockIdx.y;
  int hw0 = blockIdx.x * 64;
  int t = threadIdx.x;
  int p = t & 63;
  int c0 = t >> 6;
  const float* xb = x + (size_t)b * CIN * HW + hw0;
  #pragma unroll 4
  for (int cg = 0; cg < 64; ++cg) {
    int c = cg * 4 + c0;
    tile[p][c] = (_Float16)xb[(size_t)c * HW + p];
  }
  __syncthreads();
  _Float16* op = xp + ((size_t)b * HW + hw0) * CIN;
  #pragma unroll 4
  for (int pp = 0; pp < 64; ++pp)
    op[(size_t)pp * CIN + t] = tile[pp][t];   // 512B coalesced stores
}

// ------------------------------------------- combine experts, pack A layout --
// K order: k = r*256 + c (r = tap dh*3+dw).  A-operand (16x32 f16) lane layout
// K 0-15 -> lane m, j 0-15 ; 16-31 -> lane m+16, j 0-15 (matches B order).
__global__ __launch_bounds__(256) void pack_w_kernel(
    const float* __restrict__ ew, const float* __restrict__ eb,
    const float* __restrict__ routing, _Float16* __restrict__ wpk,
    float* __restrict__ comb_b) {
  int idx = blockIdx.x * 256 + threadIdx.x;
  int c = idx & 255;
  int m = (idx >> 8) & 511;
  int b = idx >> 17;
  float rt[NEXP];
  #pragma unroll
  for (int e = 0; e < NEXP; ++e) rt[e] = routing[b * NEXP + e];

  float acc[9] = {0, 0, 0, 0, 0, 0, 0, 0, 0};
  for (int e = 0; e < NEXP; ++e) {
    const float* w9 = ew + ((size_t)((e * COUT + m) * CIN + c)) * 9;
    float r = rt[e];
    #pragma unroll
    for (int rr = 0; rr < 9; ++rr) acc[rr] += r * w9[rr];
  }
  int mt = m >> 4, mrow = m & 15;
  int kk = c & 31;
  int blk = kk >> 3;
  int lanei = mrow + 16 * (blk >> 1);
  int j = kk & 15;
  int ktc = c >> 5;
  size_t base = (size_t)(b * 32 + mt) * KTILES;
  #pragma unroll
  for (int rr = 0; rr < 9; ++rr) {
    size_t kt = base + rr * 8 + ktc;
    wpk[kt * 512 + lanei * 16 + j] = (_Float16)acc[rr];
  }
  if (c == 0) {
    float s = 0.f;
    #pragma unroll
    for (int e = 0; e < NEXP; ++e) s += rt[e] * eb[e * COUT + m];
    comb_b[b * COUT + m] = s;
  }
}

// --------------------------------------------------- implicit-GEMM conv ------
// grid (8, 16, 16): hgroup, m-tile-pair, b.  Wave = one h row, 32 c_out x 64 w.
// Per K-tile: 2 A loads + 4 B loads (32B/lane each), 8 WMMAs -> ~21 FLOP/B.
__global__ __launch_bounds__(256) void conv_wmma_kernel(
    const _Float16* __restrict__ xp, const _Float16* __restrict__ wpk,
    const float* __restrict__ comb_b, float* __restrict__ out) {
  int hgroup = blockIdx.x;
  int mt2 = blockIdx.y;            // pair of 16-row m tiles
  int b = blockIdx.z;
  int wave = threadIdx.x >> 5;
  int lane = threadIdx.x & 31;
  int h = hgroup * 8 + wave;
  int khalf = lane >> 4, ln = lane & 15;

  const _Float16* w0 =
      wpk + (size_t)(b * 32 + mt2 * 2) * KTILES * 512 + lane * 16;
  const _Float16* w1 = w0 + (size_t)KTILES * 512;

  v8f acc[8] = {{}, {}, {}, {}, {}, {}, {}, {}};

  for (int kt = 0; kt < KTILES; ++kt) {
    int r = kt >> 3;                 // tap 0..8
    int rdiv3 = r / 3;
    int dh = rdiv3 - 1;
    int dw = (r - rdiv3 * 3) - 1;
    int cb = (kt & 7) << 5;

    v16h A0 = *(const v16h*)(w0 + (size_t)kt * 512);
    v16h A1 = *(const v16h*)(w1 + (size_t)kt * 512);
    __builtin_prefetch(w0 + (size_t)(kt + 8) * 512, 0, 0);  // global_prefetch
    __builtin_prefetch(w1 + (size_t)(kt + 8) * 512, 0, 0);

    int hh = h + dh;
    bool vh = (unsigned)hh < 64u;
    int hcl = vh ? hh : 0;
    const _Float16* xr =
        xp + ((size_t)(b * 64 + hcl) * 64) * CIN + cb + khalf * 16;

    #pragma unroll
    for (int t = 0; t < 4; ++t) {
      int ww = t * 16 + ln + dw;
      v16h Bv = {};
      if (vh && (unsigned)ww < 64u)
        Bv = *(const v16h*)(xr + (size_t)ww * CIN);   // 32B/lane contiguous
      acc[t] = __builtin_amdgcn_wmma_f32_16x16x32_f16(
          false, A0, false, Bv, (short)0, acc[t], false, false);
      acc[4 + t] = __builtin_amdgcn_wmma_f32_16x16x32_f16(
          false, A1, false, Bv, (short)0, acc[4 + t], false, false);
    }
  }

  // C/D layout: VGPR v -> M = v + 8*(lane>>4), N = lane&15
  int mhi = khalf * 8;
  #pragma unroll
  for (int mi = 0; mi < 2; ++mi) {
    #pragma unroll
    for (int v = 0; v < 8; ++v) {
      int m = (mt2 * 2 + mi) * 16 + mhi + v;
      float bi = comb_b[b * COUT + m];
      size_t o = (((size_t)b * COUT + m) * 64 + h) * 64 + ln;
      out[o]      = acc[4 * mi + 0][v] + bi;
      out[o + 16] = acc[4 * mi + 1][v] + bi;
      out[o + 32] = acc[4 * mi + 2][v] + bi;
      out[o + 48] = acc[4 * mi + 3][v] + bi;
    }
  }
}

// ------------------------------------------------------------- BN stats ------
__global__ __launch_bounds__(256) void bn_stats_kernel(
    const float* __restrict__ out, const float* __restrict__ gamma,
    const float* __restrict__ beta, float* __restrict__ scale,
    float* __restrict__ shift) {
  int m = blockIdx.x;
  float s = 0.f, q = 0.f;
  for (int i = threadIdx.x; i < B_ * HW; i += 256) {
    int b = i >> 12, hw = i & 4095;
    float v = out[(((size_t)b * COUT + m) << 12) + hw];
    s += v; q += v * v;
  }
  __shared__ float rs[256], rq[256];
  rs[threadIdx.x] = s; rq[threadIdx.x] = q;
  __syncthreads();
  for (int off = 128; off > 0; off >>= 1) {
    if (threadIdx.x < off) {
      rs[threadIdx.x] += rs[threadIdx.x + off];
      rq[threadIdx.x] += rq[threadIdx.x + off];
    }
    __syncthreads();
  }
  if (threadIdx.x == 0) {
    const float inv = 1.0f / (float)(B_ * HW);
    float mean = rs[0] * inv;
    float var = rq[0] * inv - mean * mean;
    float sc = gamma[m] * rsqrtf(var + 1e-5f);
    scale[m] = sc;
    shift[m] = beta[m] - mean * sc;
  }
}

// -------------------------------------------------- BN apply + LeakyReLU -----
__global__ __launch_bounds__(256) void bn_apply_kernel(
    float* __restrict__ out, const float* __restrict__ scale,
    const float* __restrict__ shift) {
  size_t idx = ((size_t)blockIdx.x * 256 + threadIdx.x) * 4;
  int m = (int)((idx >> 12) & 511);
  float4 v = *(float4*)(out + idx);
  float sc = scale[m], sh = shift[m];
  float r0 = v.x * sc + sh;
  float r1 = v.y * sc + sh;
  float r2 = v.z * sc + sh;
  float r3 = v.w * sc + sh;
  v.x = r0 > 0.f ? r0 : 0.1f * r0;
  v.y = r1 > 0.f ? r1 : 0.1f * r1;
  v.z = r2 > 0.f ? r2 : 0.1f * r2;
  v.w = r3 > 0.f ? r3 : 0.1f * r3;
  *(float4*)(out + idx) = v;
}

// ------------------------------------------------------------------ launch ---
extern "C" void kernel_launch(void* const* d_in, const int* in_sizes, int n_in,
                              void* d_out, int out_size, void* d_ws,
                              size_t ws_size, hipStream_t stream) {
  (void)in_sizes; (void)n_in; (void)out_size; (void)ws_size;
  const float* x     = (const float*)d_in[0];
  const float* ew    = (const float*)d_in[1];
  const float* eb    = (const float*)d_in[2];
  const float* fcw   = (const float*)d_in[3];
  const float* fcb   = (const float*)d_in[4];
  const float* gamma = (const float*)d_in[5];
  const float* beta  = (const float*)d_in[6];
  float* out = (float*)d_out;

  float* routing = (float*)d_ws;                    // 128 f32
  float* comb_b  = routing + 128;                   // 8192 f32
  float* scale   = comb_b + B_ * COUT;              // 512 f32
  float* shift   = scale + COUT;                    // 512 f32
  _Float16* xp  = (_Float16*)((char*)d_ws + 65536); // 33.5 MB
  _Float16* wpk = xp + (size_t)B_ * HW * CIN;       // 37.7 MB

  routing_kernel<<<B_, 256, 0, stream>>>(x, fcw, fcb, routing);
  pack_x_kernel<<<dim3(64, B_), 256, 0, stream>>>(x, xp);
  pack_w_kernel<<<(B_ * COUT * CIN) / 256, 256, 0, stream>>>(ew, eb, routing,
                                                             wpk, comb_b);
  conv_wmma_kernel<<<dim3(8, 16, B_), 256, 0, stream>>>(xp, wpk, comb_b, out);
  bn_stats_kernel<<<COUT, 256, 0, stream>>>(out, gamma, beta, scale, shift);
  bn_apply_kernel<<<(B_ * COUT * HW) / (256 * 4), 256, 0, stream>>>(out, scale,
                                                                    shift);
}
